// MambaExpert_77120432767223
// MI455X (gfx1250) — matrix operand order repaired and checked by reading, not compiled, not run on hardware
//
#include <hip/hip_runtime.h>
#include <math.h>

// ============================================================================
// VMamba VSS block on MI455X (gfx1250, wave32, WMMA).
//   - All dense projections run as bf16 WMMA (v_wmma_f32_16x16x32_bf16) with
//     fp32 accumulation; weights are (N,K) row-major == B^T, so fragment loads
//     are contiguous per lane with in-register fp32->bf16 conversion
//     (compiler emits global_load_b128 + v_cvt_pk_bf16_f32).
//   - Selective scan is a 3-pass chunked linear scan (96 segments x 96 steps)
//     to expose B*K*D*SEG = 147456-way parallelism instead of 1536 serial
//     recurrences. Per-block (B_t,C_t,dt) rows are staged in LDS (13.8 KB)
//     so the 96 d-channels broadcast-read instead of 96x redundant VMEM.
//     The u-stream is prefetched (global_prefetch_b8) 16 steps ahead; the
//     whole activation set is L2-resident (192 MB).
//   - Direction merge uses float atomics (index maps are involutions).
// ============================================================================

typedef __bf16 bf16;
typedef __attribute__((ext_vector_type(16))) __bf16 v16bf;
typedef __attribute__((ext_vector_type(8)))  float  v8f;

namespace {
constexpr int Bc = 4, Cc = 64, Hc = 96, Wc = 96;
constexpr int Lc  = Hc * Wc;        // 9216
constexpr int DI  = 96;             // d_inner
constexpr int NS  = 16;             // d_state
constexpr int RK  = 4;              // dt_rank
constexpr int KD  = 4;              // scan directions
constexpr int CD  = RK + 2 * NS;    // 36 (x_proj output channels)
constexpr int FFH = 128;
constexpr int SEG = 96;             // scan segments
constexpr int TSEG = Lc / SEG;      // 96 steps per segment
constexpr int MR  = Bc * Lc;        // 36864 token rows
}

// ------------------------------ math helpers -------------------------------
__device__ __forceinline__ float softplusf(float x) {
  return (x > 20.f) ? x : log1pf(__expf(x));
}
__device__ __forceinline__ float siluf(float x) { return x / (1.f + __expf(-x)); }
__device__ __forceinline__ float geluf(float x) {
  return 0.5f * x * (1.f + erff(x * 0.70710678118654752f));
}

// Direction k maps scan position t -> memory position l (involutive).
__device__ __forceinline__ int map_dir(int k, int t) {
  int tt = (k >= 2) ? (Lc - 1 - t) : t;
  if (k & 1) { int h = tt % Hc; int w = tt / Hc; return h * Wc + w; }
  return tt;
}

// --------------------------- WMMA fragment I/O -----------------------------
// A-matrix 16x32 bf16 layout (ISA 7.12.2): lanes 0-15 carry row M=lane,
// K = {k0..k0+7, k0+16..k0+23}; lanes 16-31 carry the +8 K chunks.
__device__ __forceinline__ v16bf frag_a(const float* __restrict__ rowp, int k0) {
  int lane = threadIdx.x & 31;
  const float* p = rowp + k0 + ((lane & 16) ? 8 : 0);
  v16bf f;
#pragma unroll
  for (int e = 0; e < 8; ++e) f[e] = (bf16)p[e];
#pragma unroll
  for (int e = 0; e < 8; ++e) f[8 + e] = (bf16)p[16 + e];
  return f;
}
// B-matrix 32x16 bf16 layout: lane = column N; lanes 0-15 hold
// K = {k0..k0+7, k0+16..k0+23}; lanes 16-31 carry the +8 K chunks (matches A).
// rowp points at row n of B^T (i.e. column n of B), contiguous in K.
__device__ __forceinline__ v16bf frag_b(const float* __restrict__ rowp, int k0) {
  int lane = threadIdx.x & 31;
  const float* p = rowp + k0 + ((lane & 16) ? 8 : 0);
  v16bf f;
#pragma unroll
  for (int e = 0; e < 8; ++e) f[e] = (bf16)p[e];
#pragma unroll
  for (int e = 0; e < 8; ++e) f[8 + e] = (bf16)p[16 + e];
  return f;
}
__device__ __forceinline__ v16bf frag_zero() {
  v16bf f;
#pragma unroll
  for (int e = 0; e < 16; ++e) f[e] = (bf16)0.0f;
  return f;
}
__device__ __forceinline__ v8f wmma_bf16(v16bf a, v16bf b, v8f c) {
  return __builtin_amdgcn_wmma_f32_16x16x32_bf16(false, a, false, b,
                                                 (short)0, c, false, false);
}

// ------------------------------- kernels -----------------------------------
// LN1 over C=64, reading NCHW input, writing (B*L, C) rows.
// Single fused pass for mean/sumsq (values ~N(0,1): E[x^2]-m^2 is safe).
__global__ void k_ln1(const float* __restrict__ x, const float* __restrict__ g,
                      const float* __restrict__ b, float* __restrict__ xn) {
  int row = blockIdx.x * blockDim.x + threadIdx.x;
  if (row >= MR) return;
  int bb = row / Lc, l = row % Lc;
  const float* px = x + (size_t)bb * Cc * Lc + l;
  float s = 0.f, s2 = 0.f;
  for (int c = 0; c < Cc; ++c) {
    float v = px[(size_t)c * Lc];
    s += v; s2 += v * v;
  }
  float m = s * (1.f / Cc);
  float var = s2 * (1.f / Cc) - m * m;
  float rs = rsqrtf(var + 1e-5f);
  float* po = xn + (size_t)row * Cc;
  for (int c = 0; c < Cc; ++c)
    po[c] = (px[(size_t)c * Lc] - m) * rs * g[c] + b[c];
}

// in_proj: (MR x 64) @ (192 x 64)^T -> xi (B,DI,L) for conv + z (MR,DI).
__global__ void k_inproj(const float* __restrict__ xn, const float* __restrict__ wt,
                         float* __restrict__ xi, float* __restrict__ z) {
  constexpr int NT = (2 * DI) / 16;  // 12
  int wave = blockIdx.x * (blockDim.x >> 5) + (threadIdx.x >> 5);
  int tm = wave / NT, tn = wave % NT;
  int lane = threadIdx.x & 31;
  int m0 = tm * 16, n0 = tn * 16;
  const float* arow = xn + (size_t)(m0 + (lane & 15)) * Cc;
  const float* brow = wt + (size_t)(n0 + (lane & 15)) * Cc;
  v8f acc = {};
#pragma unroll
  for (int k0 = 0; k0 < Cc; k0 += 32)
    acc = wmma_bf16(frag_a(arow, k0), frag_b(brow, k0), acc);
  int n = n0 + (lane & 15);
  int mb = m0 + ((lane & 16) ? 8 : 0);
#pragma unroll
  for (int e = 0; e < 8; ++e) {
    int m = mb + e;
    int bb = m / Lc, l = m % Lc;
    float vv = acc[e];
    if (n < DI) xi[((size_t)bb * DI + n) * Lc + l] = vv;
    else        z[(size_t)m * DI + (n - DI)] = vv;
  }
}

// Depthwise 3x3 SAME conv + bias + SiLU; writes channels-last xs (B,L,DI).
__global__ void k_conv(const float* __restrict__ xi, const float* __restrict__ cw,
                       const float* __restrict__ cb, float* __restrict__ xs) {
  int idx = blockIdx.x * blockDim.x + threadIdx.x;  // B*DI*L threads
  int l = idx % Lc; int t = idx / Lc; int d = t % DI; int bb = t / DI;
  int h = l / Wc, w = l % Wc;
  const float* pl = xi + ((size_t)bb * DI + d) * Lc;
  const float* kw = cw + (size_t)d * 9;
  float s = cb[d];
#pragma unroll
  for (int dy = -1; dy <= 1; ++dy) {
    int hh = h + dy; if (hh < 0 || hh >= Hc) continue;
#pragma unroll
    for (int dx = -1; dx <= 1; ++dx) {
      int ww = w + dx; if (ww < 0 || ww >= Wc) continue;
      s += kw[(dy + 1) * 3 + (dx + 1)] * pl[hh * Wc + ww];
    }
  }
  xs[((size_t)bb * Lc + l) * DI + d] = siluf(s);
}

// x_proj: per (b,k): C[t,c] = sum_d xs[b, map(k,t), d] * x_proj_w[k,c,d].
// M = L (rows permuted per direction), N = 36 (3 tiles, guarded), K = 96.
__global__ void k_xproj(const float* __restrict__ xs, const float* __restrict__ xpw,
                        float* __restrict__ dbl) {
  constexpr int MT = Lc / 16;  // 576
  constexpr int NT = 3;        // covers 48 cols, guard at 36
  int wave = blockIdx.x * (blockDim.x >> 5) + (threadIdx.x >> 5);
  int tn = wave % NT; int tm = (wave / NT) % MT; int bk = wave / (NT * MT);
  int k = bk % KD, bb = bk / KD;
  int lane = threadIdx.x & 31;
  int m0 = tm * 16, n0 = tn * 16;
  const float* arow = xs + ((size_t)bb * Lc + map_dir(k, m0 + (lane & 15))) * DI;
  int nr = n0 + (lane & 15);
  bool bok = (nr < CD);
  const float* brow = xpw + ((size_t)k * CD + (bok ? nr : 0)) * DI;
  v8f acc = {};
#pragma unroll
  for (int k0 = 0; k0 < DI; k0 += 32) {
    v16bf bf_ = bok ? frag_b(brow, k0) : frag_zero();
    acc = wmma_bf16(frag_a(arow, k0), bf_, acc);
  }
  int c = n0 + (lane & 15);
  if (c >= CD) return;
  int mb = m0 + ((lane & 16) ? 8 : 0);
  float* po = dbl + (size_t)bk * Lc * CD;
#pragma unroll
  for (int e = 0; e < 8; ++e) po[(size_t)(mb + e) * CD + c] = acc[e];
}

// Scan pass 1: per (b,k,d,seg): segment product of a_t and state with h0=0.
// The 96 x 36 float (dt,B,C) rows for this segment are staged in LDS once;
// every step then broadcast-reads from LDS instead of 96x redundant VMEM.
__global__ void k_scan1(const float* __restrict__ dbl, const float* __restrict__ xs,
                        const float* __restrict__ dtw, const float* __restrict__ dtb,
                        const float* __restrict__ alog,
                        float* __restrict__ segA, float* __restrict__ segH) {
  __shared__ float sdbl[TSEG * CD];  // 13824 B of the 320 KB WGP LDS
  int seg = blockIdx.x % SEG; int bk = blockIdx.x / SEG;
  int k = bk % KD, bb = bk / KD;
  int d = threadIdx.x;
  const float* db = dbl + ((size_t)bk * Lc + (size_t)seg * TSEG) * CD;
  for (int i = d; i < TSEG * CD; i += DI) sdbl[i] = db[i];
  __syncthreads();

  float a[NS], h[NS], ap[NS];
#pragma unroll
  for (int n = 0; n < NS; ++n) {
    a[n] = -__expf(alog[((size_t)k * DI + d) * NS + n]);
    h[n] = 0.f; ap[n] = 1.f;
  }
  const float* wr = dtw + ((size_t)k * DI + d) * RK;
  float w0 = wr[0], w1 = wr[1], w2 = wr[2], w3 = wr[3];
  float bia = dtb[(size_t)k * DI + d];
  const float* xb = xs + (size_t)bb * Lc * DI;
  for (int tt = 0; tt < TSEG; ++tt) {
    int t = seg * TSEG + tt;
    if (tt + 16 < TSEG)  // prefetch u-stream (L2-resident) 16 steps ahead
      __builtin_prefetch(xb + (size_t)map_dir(k, t + 16) * DI + d, 0, 1);
    const float* row = sdbl + tt * CD;
    float dl = softplusf(bia + w0 * row[0] + w1 * row[1] + w2 * row[2] + w3 * row[3]);
    float du = dl * xb[(size_t)map_dir(k, t) * DI + d];
#pragma unroll
    for (int n = 0; n < NS; ++n) {
      float ex = __expf(dl * a[n]);
      h[n] = ex * h[n] + du * row[RK + n];
      ap[n] *= ex;
    }
  }
  size_t base = (((size_t)bk * DI + d) * SEG + seg) * NS;
#pragma unroll
  for (int n = 0; n < NS; ++n) { segA[base + n] = ap[n]; segH[base + n] = h[n]; }
}

// Scan pass 2: serial inter-segment scan per (b,k,d,n); rewrites segH with
// the exclusive carry-in state h0 of each segment.
__global__ void k_scan2(const float* __restrict__ segA, float* __restrict__ segH) {
  int idx = blockIdx.x * blockDim.x + threadIdx.x;  // B*K*DI*NS = 24576
  int n = idx & (NS - 1);
  int chan = idx >> 4;  // bk*DI + d
  size_t base = (size_t)chan * SEG * NS + n;
  float carry = 0.f;
  for (int s = 0; s < SEG; ++s) {
    size_t o = base + (size_t)s * NS;
    float hend = segH[o];
    segH[o] = carry;
    carry = segA[o] * carry + hend;
  }
}

// Scan pass 3: replay with correct h0, emit y_t = C_t.h_t + u*D, scatter-add
// into the spatial accumulator across the 4 directions.
__global__ void k_scan3(const float* __restrict__ dbl, const float* __restrict__ xs,
                        const float* __restrict__ dtw, const float* __restrict__ dtb,
                        const float* __restrict__ alog, const float* __restrict__ Ds,
                        const float* __restrict__ segH, float* __restrict__ y) {
  __shared__ float sdbl[TSEG * CD];
  int seg = blockIdx.x % SEG; int bk = blockIdx.x / SEG;
  int k = bk % KD, bb = bk / KD;
  int d = threadIdx.x;
  const float* db = dbl + ((size_t)bk * Lc + (size_t)seg * TSEG) * CD;
  for (int i = d; i < TSEG * CD; i += DI) sdbl[i] = db[i];
  __syncthreads();

  float a[NS], h[NS];
  size_t hb = (((size_t)bk * DI + d) * SEG + seg) * NS;
#pragma unroll
  for (int n = 0; n < NS; ++n) {
    a[n] = -__expf(alog[((size_t)k * DI + d) * NS + n]);
    h[n] = segH[hb + n];
  }
  const float* wr = dtw + ((size_t)k * DI + d) * RK;
  float w0 = wr[0], w1 = wr[1], w2 = wr[2], w3 = wr[3];
  float bia = dtb[(size_t)k * DI + d];
  float dd = Ds[(size_t)k * DI + d];
  const float* xb = xs + (size_t)bb * Lc * DI;
  for (int tt = 0; tt < TSEG; ++tt) {
    int t = seg * TSEG + tt;
    if (tt + 16 < TSEG)
      __builtin_prefetch(xb + (size_t)map_dir(k, t + 16) * DI + d, 0, 1);
    const float* row = sdbl + tt * CD;
    float dl = softplusf(bia + w0 * row[0] + w1 * row[1] + w2 * row[2] + w3 * row[3]);
    int lm = map_dir(k, t);
    float u = xb[(size_t)lm * DI + d];
    float du = dl * u;
    float yv = 0.f;
#pragma unroll
    for (int n = 0; n < NS; ++n) {
      float ex = __expf(dl * a[n]);
      h[n] = ex * h[n] + du * row[RK + n];
      yv += row[RK + NS + n] * h[n];
    }
    atomicAdd(&y[((size_t)bb * Lc + lm) * DI + d], yv + u * dd);
  }
}

// out_norm (LN over DI) * silu(z), single fused stats pass.
__global__ void k_outnorm(const float* __restrict__ y, const float* __restrict__ z,
                          const float* __restrict__ g, const float* __restrict__ b,
                          float* __restrict__ yn) {
  int row = blockIdx.x * blockDim.x + threadIdx.x;
  if (row >= MR) return;
  const float* py = y + (size_t)row * DI;
  const float* pz = z + (size_t)row * DI;
  float s = 0.f, s2 = 0.f;
  for (int d = 0; d < DI; ++d) { float v = py[d]; s += v; s2 += v * v; }
  float m = s * (1.f / DI);
  float var = s2 * (1.f / DI) - m * m;
  float rs = rsqrtf(var + 1e-5f);
  float* po = yn + (size_t)row * DI;
  for (int d = 0; d < DI; ++d)
    po[d] = ((py[d] - m) * rs * g[d] + b[d]) * siluf(pz[d]);
}

// out_proj + residual with original NCHW x: x2[m,c] = x + yn @ W^T
__global__ void k_outproj(const float* __restrict__ yn, const float* __restrict__ wt,
                          const float* __restrict__ x, float* __restrict__ x2) {
  constexpr int NT = Cc / 16;  // 4
  int wave = blockIdx.x * (blockDim.x >> 5) + (threadIdx.x >> 5);
  int tm = wave / NT, tn = wave % NT;
  int lane = threadIdx.x & 31;
  int m0 = tm * 16, n0 = tn * 16;
  const float* arow = yn + (size_t)(m0 + (lane & 15)) * DI;
  const float* brow = wt + (size_t)(n0 + (lane & 15)) * DI;
  v8f acc = {};
#pragma unroll
  for (int k0 = 0; k0 < DI; k0 += 32)
    acc = wmma_bf16(frag_a(arow, k0), frag_b(brow, k0), acc);
  int n = n0 + (lane & 15);
  int mb = m0 + ((lane & 16) ? 8 : 0);
#pragma unroll
  for (int e = 0; e < 8; ++e) {
    int m = mb + e;
    int bb = m / Lc, l = m % Lc;
    x2[(size_t)m * Cc + n] = acc[e] + x[((size_t)bb * Cc + n) * Lc + l];
  }
}

// LN over contiguous rows of width Cc (for LN2), fused stats pass.
__global__ void k_lnrow(const float* __restrict__ in, const float* __restrict__ g,
                        const float* __restrict__ b, float* __restrict__ out) {
  int row = blockIdx.x * blockDim.x + threadIdx.x;
  if (row >= MR) return;
  const float* pi = in + (size_t)row * Cc;
  float s = 0.f, s2 = 0.f;
  for (int c = 0; c < Cc; ++c) { float v = pi[c]; s += v; s2 += v * v; }
  float m = s * (1.f / Cc);
  float var = s2 * (1.f / Cc) - m * m;
  float rs = rsqrtf(var + 1e-5f);
  float* po = out + (size_t)row * Cc;
  for (int c = 0; c < Cc; ++c) po[c] = (pi[c] - m) * rs * g[c] + b[c];
}

// FFN GEMM 1: (MR x 64) @ (128 x 64)^T, epilogue bias + exact GELU.
__global__ void k_ffn1(const float* __restrict__ xn2, const float* __restrict__ w1,
                       const float* __restrict__ b1, float* __restrict__ hge) {
  constexpr int NT = FFH / 16;  // 8
  int wave = blockIdx.x * (blockDim.x >> 5) + (threadIdx.x >> 5);
  int tm = wave / NT, tn = wave % NT;
  int lane = threadIdx.x & 31;
  int m0 = tm * 16, n0 = tn * 16;
  const float* arow = xn2 + (size_t)(m0 + (lane & 15)) * Cc;
  const float* brow = w1 + (size_t)(n0 + (lane & 15)) * Cc;
  v8f acc = {};
#pragma unroll
  for (int k0 = 0; k0 < Cc; k0 += 32)
    acc = wmma_bf16(frag_a(arow, k0), frag_b(brow, k0), acc);
  int n = n0 + (lane & 15);
  int mb = m0 + ((lane & 16) ? 8 : 0);
  float bias = b1[n];
#pragma unroll
  for (int e = 0; e < 8; ++e)
    hge[(size_t)(mb + e) * FFH + n] = geluf(acc[e] + bias);
}

// FFN GEMM 2 + bias + residual, writing the final NCHW output.
__global__ void k_ffn2(const float* __restrict__ hge, const float* __restrict__ w2,
                       const float* __restrict__ b2, const float* __restrict__ x2,
                       float* __restrict__ out) {
  constexpr int NT = Cc / 16;  // 4
  int wave = blockIdx.x * (blockDim.x >> 5) + (threadIdx.x >> 5);
  int tm = wave / NT, tn = wave % NT;
  int lane = threadIdx.x & 31;
  int m0 = tm * 16, n0 = tn * 16;
  const float* arow = hge + (size_t)(m0 + (lane & 15)) * FFH;
  const float* brow = w2 + (size_t)(n0 + (lane & 15)) * FFH;
  v8f acc = {};
#pragma unroll
  for (int k0 = 0; k0 < FFH; k0 += 32)
    acc = wmma_bf16(frag_a(arow, k0), frag_b(brow, k0), acc);
  int n = n0 + (lane & 15);
  int mb = m0 + ((lane & 16) ? 8 : 0);
  float bias = b2[n];
#pragma unroll
  for (int e = 0; e < 8; ++e) {
    int m = mb + e;
    int bb = m / Lc, l = m % Lc;
    out[((size_t)bb * Cc + n) * Lc + l] = acc[e] + bias + x2[(size_t)m * Cc + n];
  }
}

// ------------------------------- launcher ----------------------------------
extern "C" void kernel_launch(void* const* d_in, const int* in_sizes, int n_in,
                              void* d_out, int out_size, void* d_ws, size_t ws_size,
                              hipStream_t stream) {
  (void)in_sizes; (void)n_in; (void)out_size; (void)ws_size;
  const float* x    = (const float*)d_in[0];
  const float* n1g  = (const float*)d_in[1];
  const float* n1b  = (const float*)d_in[2];
  const float* ipw  = (const float*)d_in[3];
  const float* cw   = (const float*)d_in[4];
  const float* cb   = (const float*)d_in[5];
  const float* xpw  = (const float*)d_in[6];
  const float* dtw  = (const float*)d_in[7];
  const float* dtb  = (const float*)d_in[8];
  const float* alog = (const float*)d_in[9];
  const float* Dsp  = (const float*)d_in[10];
  const float* ong  = (const float*)d_in[11];
  const float* onb  = (const float*)d_in[12];
  const float* opw  = (const float*)d_in[13];
  const float* n2g  = (const float*)d_in[14];
  const float* n2b  = (const float*)d_in[15];
  const float* fw1  = (const float*)d_in[16];
  const float* fb1  = (const float*)d_in[17];
  const float* fw2  = (const float*)d_in[18];
  const float* fb2  = (const float*)d_in[19];
  float* out = (float*)d_out;

  // Workspace carve-out (~106 MB), with lifetime-based aliasing.
  char* wp = (char*)d_ws;
  auto alloc = [&](size_t nfl) {
    float* p = (float*)wp;
    wp += ((nfl * sizeof(float) + 255) & ~(size_t)255);
    return p;
  };
  float* xn   = alloc((size_t)MR * Cc);                 // LN1 out; reused as x2
  float* xi   = alloc((size_t)Bc * DI * Lc);            // conv in; reused as yn
  float* z    = alloc((size_t)MR * DI);                 // gate branch
  float* xs   = alloc((size_t)MR * DI);                 // conv+silu (B,L,DI)
  float* dbl  = alloc((size_t)Bc * KD * Lc * CD);       // x_proj out; reused hge
  float* segA = alloc((size_t)Bc * KD * DI * SEG * NS); // seg prod; reused xn2
  float* segH = alloc((size_t)Bc * KD * DI * SEG * NS); // seg state -> seg h0
  float* yac  = alloc((size_t)MR * DI);                 // direction accumulator
  float* x2 = xn; float* yn = xi; float* xn2 = segA; float* hge = dbl;

  k_ln1   <<<MR / 256, 256, 0, stream>>>(x, n1g, n1b, xn);
  k_inproj<<<(MR / 16) * ((2 * DI) / 16) / 8, 256, 0, stream>>>(xn, ipw, xi, z);
  k_conv  <<<(Bc * DI * Lc) / 256, 256, 0, stream>>>(xi, cw, cb, xs);
  k_xproj <<<(Bc * KD * (Lc / 16) * 3) / 8, 256, 0, stream>>>(xs, xpw, dbl);
  k_scan1 <<<Bc * KD * SEG, DI, 0, stream>>>(dbl, xs, dtw, dtb, alog, segA, segH);
  k_scan2 <<<(Bc * KD * DI * NS) / 256, 256, 0, stream>>>(segA, segH);
  hipMemsetAsync(yac, 0, (size_t)MR * DI * sizeof(float), stream);
  k_scan3 <<<Bc * KD * SEG, DI, 0, stream>>>(dbl, xs, dtw, dtb, alog, Dsp, segH, yac);
  k_outnorm<<<MR / 256, 256, 0, stream>>>(yac, z, ong, onb, yn);
  k_outproj<<<(MR / 16) * (Cc / 16) / 8, 256, 0, stream>>>(yn, opw, x, x2);
  k_lnrow <<<MR / 256, 256, 0, stream>>>(x2, n2g, n2b, xn2);
  k_ffn1  <<<(MR / 16) * (FFH / 16) / 8, 256, 0, stream>>>(xn2, fw1, fb1, hge);
  k_ffn2  <<<(MR / 16) * (Cc / 16) / 8, 256, 0, stream>>>(hge, fw2, fb2, x2, out);
}
